// GraphEncoder_GAT_wSKIP_10917806866971
// MI455X (gfx1250) — matrix operand order repaired and checked by reading, not compiled, not run on hardware
//
#include <hip/hip_runtime.h>
#include <hip/hip_bf16.h>

typedef __attribute__((ext_vector_type(16))) _Float16 v16h;
typedef __attribute__((ext_vector_type(8)))  _Float16 v8h;
typedef __attribute__((ext_vector_type(8)))  float    v8f;

#define GAT_H 4
#define GAT_C 64
#define GAT_F 256   // H*C == F_IN
#define GAT_G 64
#define NHID 1024
#define NOUT 768

// ---------------- utility ----------------

__global__ void zero_kernel(float* __restrict__ p, int n) {
  int i = blockIdx.x * blockDim.x + threadIdx.x;
  if (i < n) p[i] = 0.0f;
}

__global__ void f32_to_f16_kernel(const float* __restrict__ src,
                                  _Float16* __restrict__ dst, int n) {
  int i = blockIdx.x * blockDim.x + threadIdx.x;
  if (i < n) dst[i] = (_Float16)src[i];
}

// monotonic encode/decode so unsigned atomicMax == float max (handles negatives)
__device__ __forceinline__ unsigned enc_f(float f) {
  unsigned u = __float_as_uint(f);
  return (u & 0x80000000u) ? ~u : (u | 0x80000000u);
}
__device__ __forceinline__ float dec_f(unsigned e) {
  return (e & 0x80000000u) ? __uint_as_float(e & 0x7FFFFFFFu)
                           : __uint_as_float(~e);
}

// ---------------- WMMA GEMM: Cout[M,Nc] = act(Ah[M,K] @ Bh[K,Nc] + bias) ----------------
// A and B are pre-converted f16. Grid: (M/16, Nc/128). Block: 256 threads = 8 waves;
// wave w owns columns [blockIdx.y*128 + w*16, +16). A 16-row panel staged in LDS.

__global__ void __launch_bounds__(256)
gemm_wmma(const _Float16* __restrict__ A, const _Float16* __restrict__ B,
          const float* __restrict__ bias, float* __restrict__ Cout,
          int K, int Nc, int act, int useBias) {
  __shared__ _Float16 Alds[16 * 1024];   // supports K up to 1024 (32 KB)

  const int tid  = threadIdx.x;
  const int wave = tid >> 5;
  const int lane = tid & 31;
  const int row0 = blockIdx.x * 16;
  const int col0 = blockIdx.y * 128 + wave * 16;

  // cooperative stage: A[row0..row0+15, 0..K) -> LDS (row-major, stride K), 8 halfs/copy
  const int chunks = (16 * K) >> 3;
  for (int idx = tid; idx < chunks; idx += 256) {
    int flat = idx << 3;
    int r = flat / K, k = flat - r * K;
    *reinterpret_cast<v8h*>(Alds + flat) =
        *reinterpret_cast<const v8h*>(A + (size_t)(row0 + r) * K + k);
  }
  __syncthreads();

  v8f c = {};
  // A fragment layout (16-bit A, 16x32): lanes 0-15 -> row M=lane, K chunks {0..7,16..23};
  // lanes 16-31 -> row M=lane-16, K chunks {8..15,24..31}.
  const int rA   = lane & 15;
  const int kOff = (lane < 16) ? 0 : 8;

  for (int k0 = 0; k0 < K; k0 += 32) {
    const _Float16* arow = Alds + rA * K + k0 + kOff;
    v8h alo = *reinterpret_cast<const v8h*>(arow);
    v8h ahi = *reinterpret_cast<const v8h*>(arow + 16);
    v16h a = __builtin_shufflevector(alo, ahi, 0,1,2,3,4,5,6,7,8,9,10,11,12,13,14,15);

    // B fragment: lane holds column N = col0+lane%16, K chunks same as A ({0..7,16..23} / {8..15,24..31})
    const _Float16* bp = B + (size_t)(k0 + kOff) * Nc + col0 + (lane & 15);
    v16h b;
#pragma unroll
    for (int s = 0; s < 16; ++s) b[s] = bp[(size_t)(s < 8 ? s : s + 8) * Nc];

    c = __builtin_amdgcn_wmma_f32_16x16x32_f16(false, a, false, b,
                                               (short)0, c, false, false);
  }

  // C/D layout: VGPR i -> M = i (lanes 0-15) or i+8 (lanes 16-31); N = lane%16
  const int col   = col0 + (lane & 15);
  const int rbase = row0 + ((lane < 16) ? 0 : 8);
  const float bv  = useBias ? bias[col] : 0.0f;
#pragma unroll
  for (int i = 0; i < 8; ++i) {
    float v = c[i] + bv;
    if (act) v = fmaxf(v, 0.0f);
    Cout[(size_t)(rbase + i) * Nc + col] = v;
  }
}

// ---------------- attention scalars: alpha_src/dst[n,h] = <h[n,h,:], a[h,:]> ----------------
// grid = N blocks, 128 threads (4 waves, one head per wave)

__global__ void __launch_bounds__(128)
alphas_kernel(const float* __restrict__ h, const float* __restrict__ as_,
              const float* __restrict__ ad_, float* __restrict__ asrc,
              float* __restrict__ adst) {
  const int n    = blockIdx.x;
  const int hh   = threadIdx.x >> 5;
  const int lane = threadIdx.x & 31;
  const float* hp  = h   + (size_t)n * GAT_F + hh * GAT_C;
  const float* asp = as_ + hh * GAT_C;
  const float* adp = ad_ + hh * GAT_C;
  float h0 = hp[lane], h1 = hp[lane + 32];
  float s0 = h0 * asp[lane] + h1 * asp[lane + 32];
  float s1 = h0 * adp[lane] + h1 * adp[lane + 32];
#pragma unroll
  for (int o = 16; o > 0; o >>= 1) {
    s0 += __shfl_down(s0, o, 32);
    s1 += __shfl_down(s1, o, 32);
  }
  if (lane == 0) {
    asrc[n * GAT_H + hh] = s0;
    adst[n * GAT_H + hh] = s1;
  }
}

// ---------------- edge passes ----------------

__device__ __forceinline__ void edge_nodes(const int* __restrict__ src,
                                           const int* __restrict__ dst,
                                           int E, int e, int& s, int& d) {
  if (e < E) { s = src[e]; d = dst[e]; } else { s = d = e - E; }
}

__global__ void edge_max_kernel(const int* __restrict__ src, const int* __restrict__ dst,
                                int E, int EtH,
                                const float* __restrict__ asrc,
                                const float* __restrict__ adst,
                                unsigned* __restrict__ menc) {
  int idx = blockIdx.x * blockDim.x + threadIdx.x;
  if (idx >= EtH) return;
  int e = idx >> 2, hh = idx & 3, s, d;
  edge_nodes(src, dst, E, e, s, d);
  float ev = asrc[s * GAT_H + hh] + adst[d * GAT_H + hh];
  ev = (ev > 0.0f) ? ev : 0.2f * ev;
  atomicMax(&menc[d * GAT_H + hh], enc_f(ev));
}

__global__ void edge_expsum_kernel(const int* __restrict__ src, const int* __restrict__ dst,
                                   int E, int EtH,
                                   const float* __restrict__ asrc,
                                   const float* __restrict__ adst,
                                   const unsigned* __restrict__ menc,
                                   float* __restrict__ denom,
                                   float* __restrict__ eexp) {
  int idx = blockIdx.x * blockDim.x + threadIdx.x;
  if (idx >= EtH) return;
  int e = idx >> 2, hh = idx & 3, s, d;
  edge_nodes(src, dst, E, e, s, d);
  float ev = asrc[s * GAT_H + hh] + adst[d * GAT_H + hh];
  ev = (ev > 0.0f) ? ev : 0.2f * ev;
  float m  = dec_f(menc[d * GAT_H + hh]);
  float ex = __expf(ev - m);
  eexp[idx] = ex;
  atomicAdd(&denom[d * GAT_H + hh], ex);
}

// one block per edge, 256 threads = all H*C channels
__global__ void __launch_bounds__(256)
edge_agg_kernel(const int* __restrict__ src, const int* __restrict__ dst, int E,
                const float* __restrict__ hlin, const float* __restrict__ eexp,
                const float* __restrict__ denom, float* __restrict__ out) {
  int e = blockIdx.x, cch = threadIdx.x, hh = cch >> 6, s, d;
  edge_nodes(src, dst, E, e, s, d);
  float w = eexp[e * GAT_H + hh] / denom[d * GAT_H + hh];
  atomicAdd(&out[(size_t)d * GAT_F + cch], hlin[(size_t)s * GAT_F + cch] * w);
}

// bias + activation, plus f16 shadow copy (A input of next WMMA GEMM)
__global__ void bias_act_kernel(float* __restrict__ buf, const float* __restrict__ bias,
                                _Float16* __restrict__ shadow, int total, int act) {
  int i = blockIdx.x * blockDim.x + threadIdx.x;
  if (i >= total) return;
  float v = buf[i] + bias[i & (GAT_F - 1)];
  if (act) v = fmaxf(v, 0.0f);
  buf[i]    = v;
  shadow[i] = (_Float16)v;
}

// ---------------- pooling ----------------

__global__ void pool_sum_kernel(const float* __restrict__ h, const int* __restrict__ batch,
                                float* __restrict__ sums, int total) {
  int i = blockIdx.x * blockDim.x + threadIdx.x;
  if (i >= total) return;
  int n = i >> 8, cc = i & 255;
  atomicAdd(&sums[batch[n] * GAT_F + cc], h[i]);
}

__global__ void pool_count_kernel(const int* __restrict__ batch, float* __restrict__ counts,
                                  int N) {
  int n = blockIdx.x * blockDim.x + threadIdx.x;
  if (n < N) atomicAdd(&counts[batch[n]], 1.0f);
}

__global__ void pool_div_kernel(const float* __restrict__ sums,
                                const float* __restrict__ counts,
                                float* __restrict__ pooled) {
  int i = blockIdx.x * blockDim.x + threadIdx.x;
  if (i >= GAT_G * GAT_F) return;
  pooled[i] = sums[i] / fmaxf(counts[i >> 8], 1.0f);
}

// ---------------- launch ----------------

extern "C" void kernel_launch(void* const* d_in, const int* in_sizes, int n_in,
                              void* d_out, int out_size, void* d_ws, size_t ws_size,
                              hipStream_t stream) {
  (void)n_in; (void)out_size; (void)ws_size;
  const float* x     = (const float*)d_in[0];
  const int*   ei    = (const int*)d_in[1];
  const int*   batch = (const int*)d_in[2];
  const float* W1 = (const float*)d_in[3],  *as1 = (const float*)d_in[4];
  const float* ad1 = (const float*)d_in[5], *b1  = (const float*)d_in[6];
  const float* W2 = (const float*)d_in[7],  *as2 = (const float*)d_in[8];
  const float* ad2 = (const float*)d_in[9], *b2  = (const float*)d_in[10];
  const float* W3 = (const float*)d_in[11], *as3 = (const float*)d_in[12];
  const float* ad3 = (const float*)d_in[13], *b3 = (const float*)d_in[14];
  const float* Wm1 = (const float*)d_in[15], *bm1 = (const float*)d_in[16];
  const float* Wm2 = (const float*)d_in[17], *bm2 = (const float*)d_in[18];

  const int N  = in_sizes[2];        // batch has N entries
  const int E  = in_sizes[1] / 2;    // edge_index is [2,E]
  const int Et = E + N;              // + self loops
  const int* srcI = ei;
  const int* dstI = ei + E;

  float* ws = (float*)d_ws;
  size_t off = 0;
  auto alloc = [&](size_t n) { float* p = ws + off; off += (n + 63) & ~(size_t)63; return p; };
  auto alloc_h = [&](size_t nh2) { return (_Float16*)alloc((nh2 + 1) / 2); };

  float*    buf0   = alloc((size_t)N * GAT_F);              // hlin (f32)
  float*    buf1   = alloc((size_t)N * GAT_F);              // aggregated output (f32)
  _Float16* nodeh  = alloc_h((size_t)N * GAT_F);            // f16 A input (ping-pong shadow)
  float*    asrc   = alloc((size_t)N * GAT_H);
  float*    adst   = alloc((size_t)N * GAT_H);
  unsigned* menc   = (unsigned*)alloc((size_t)N * GAT_H);
  float*    denom  = alloc((size_t)N * GAT_H);
  float*    eexp   = alloc((size_t)Et * GAT_H);
  float*    sums   = alloc(GAT_G * GAT_F);
  float*    counts = alloc(GAT_G);
  float*    pooled = alloc(GAT_G * GAT_F);
  float*    hid    = alloc(GAT_G * NHID);
  _Float16* W1h    = alloc_h(GAT_F * GAT_F);
  _Float16* W2h    = alloc_h(GAT_F * GAT_F);
  _Float16* W3h    = alloc_h(GAT_F * GAT_F);
  _Float16* Wm1h   = alloc_h(GAT_F * NHID);
  _Float16* Wm2h   = alloc_h(NHID * NOUT);
  _Float16* pooledh= alloc_h(GAT_G * GAT_F);
  _Float16* hidh   = alloc_h(GAT_G * NHID);

  const int nf = N * GAT_F;
  const int nh = N * GAT_H;
  const int eh = Et * GAT_H;
  auto cdiv = [](int a, int b) { return (a + b - 1) / b; };

  // one-time (per launch) f16 conversions of weights and input features
  f32_to_f16_kernel<<<cdiv(nf, 256), 256, 0, stream>>>(x, nodeh, nf);
  f32_to_f16_kernel<<<cdiv(GAT_F * GAT_F, 256), 256, 0, stream>>>(W1, W1h, GAT_F * GAT_F);
  f32_to_f16_kernel<<<cdiv(GAT_F * GAT_F, 256), 256, 0, stream>>>(W2, W2h, GAT_F * GAT_F);
  f32_to_f16_kernel<<<cdiv(GAT_F * GAT_F, 256), 256, 0, stream>>>(W3, W3h, GAT_F * GAT_F);
  f32_to_f16_kernel<<<cdiv(GAT_F * NHID, 256), 256, 0, stream>>>(Wm1, Wm1h, GAT_F * NHID);
  f32_to_f16_kernel<<<cdiv(NHID * NOUT, 256), 256, 0, stream>>>(Wm2, Wm2h, NHID * NOUT);

  auto run_layer = [&](const _Float16* Wh, const float* as_, const float* ad_,
                       const float* b_, int act) {
    gemm_wmma<<<dim3(N / 16, GAT_F / 128), 256, 0, stream>>>(
        nodeh, Wh, nullptr, buf0, GAT_F, GAT_F, 0, 0);
    alphas_kernel<<<N, 128, 0, stream>>>(buf0, as_, ad_, asrc, adst);
    zero_kernel<<<cdiv(nh, 256), 256, 0, stream>>>((float*)menc, nh);
    zero_kernel<<<cdiv(nh, 256), 256, 0, stream>>>(denom, nh);
    edge_max_kernel<<<cdiv(eh, 256), 256, 0, stream>>>(srcI, dstI, E, eh, asrc, adst, menc);
    edge_expsum_kernel<<<cdiv(eh, 256), 256, 0, stream>>>(srcI, dstI, E, eh, asrc, adst,
                                                          menc, denom, eexp);
    zero_kernel<<<cdiv(nf, 256), 256, 0, stream>>>(buf1, nf);
    edge_agg_kernel<<<Et, 256, 0, stream>>>(srcI, dstI, E, buf0, eexp, denom, buf1);
    bias_act_kernel<<<cdiv(nf, 256), 256, 0, stream>>>(buf1, b_, nodeh, nf, act);
  };

  run_layer(W1h, as1, ad1, b1, 1);
  run_layer(W2h, as2, ad2, b2, 1);
  run_layer(W3h, as3, ad3, b3, 0);

  // global mean pool (reads f32 buf1 for accuracy)
  zero_kernel<<<cdiv(GAT_G * GAT_F, 256), 256, 0, stream>>>(sums, GAT_G * GAT_F);
  zero_kernel<<<1, GAT_G, 0, stream>>>(counts, GAT_G);
  pool_sum_kernel<<<cdiv(nf, 256), 256, 0, stream>>>(buf1, batch, sums, nf);
  pool_count_kernel<<<cdiv(N, 256), 256, 0, stream>>>(batch, counts, N);
  pool_div_kernel<<<cdiv(GAT_G * GAT_F, 256), 256, 0, stream>>>(sums, counts, pooled);

  // MLP head (WMMA)
  f32_to_f16_kernel<<<cdiv(GAT_G * GAT_F, 256), 256, 0, stream>>>(pooled, pooledh,
                                                                  GAT_G * GAT_F);
  gemm_wmma<<<dim3(GAT_G / 16, NHID / 128), 256, 0, stream>>>(
      pooledh, Wm1h, bm1, hid, GAT_F, NHID, 1, 1);
  f32_to_f16_kernel<<<cdiv(GAT_G * NHID, 256), 256, 0, stream>>>(hid, hidh, GAT_G * NHID);
  gemm_wmma<<<dim3(GAT_G / 16, NOUT / 128), 256, 0, stream>>>(
      hidh, Wm2h, bm2, (float*)d_out, NHID, NOUT, 0, 1);
}
